// SpanTransformer_59115929862327
// MI455X (gfx1250) — matrix operand order repaired and checked by reading, not compiled, not run on hardware
//
#include <hip/hip_runtime.h>
#include <hip/hip_bf16.h>
#include <math.h>

// ---------------- problem constants (from reference) ----------------
#define B_    2
#define S_    300
#define D_    512
#define DS_   256
#define H_    8
#define L_    3
#define FF_   2048
#define NE_   20
#define NT_   34
#define NR_   10
#define DE_   64
#define DH_   64          // D/H
#define PHID_ 512
#define F_    (2*DS_+DE_) // 576 pair feature dim
#define PP_   (S_*(S_-1)) // 89700 pairs per batch
#define P_    (B_*PP_)    // 179400 total pairs
#define MR_   (B_*S_)     // 600 token rows

typedef __attribute__((ext_vector_type(16))) _Float16 v16h;
typedef __attribute__((ext_vector_type(8)))  _Float16 v8h;
typedef __attribute__((ext_vector_type(8)))  float    v8f;
typedef __attribute__((ext_vector_type(4)))  float    v4f;

// async-load wait (ASYNCcnt): builtin if the toolchain has it, else raw SOPP
static __device__ __forceinline__ void wait_async0() {
#if defined(__has_builtin)
#if __has_builtin(__builtin_amdgcn_s_wait_asynccnt)
    __builtin_amdgcn_s_wait_asynccnt(0);
    return;
#endif
#endif
    asm volatile("s_wait_asynccnt 0x0" ::: "memory");
}

// one 16B global->LDS async copy (GLOBAL_LOAD_ASYNC_TO_LDS_B128, GV mode)
static __device__ __forceinline__ void async_copy_b128(const void* gptr, void* lds_ptr) {
    unsigned lds_off = (unsigned)(size_t)lds_ptr;           // addr[31:0] == LDS offset
    unsigned long long ga = (unsigned long long)(size_t)gptr;
    asm volatile("global_load_async_to_lds_b128 %0, %1, off"
                 :: "v"(lds_off), "v"(ga) : "memory");
}

// =====================================================================
// f32 -> f16 tensor conversion (weights are converted once per call)
// =====================================================================
__global__ __launch_bounds__(256)
void cvt_f32_f16_kernel(const float* __restrict__ src, _Float16* __restrict__ dst, int n)
{
    int i = blockIdx.x * 256 + threadIdx.x;
    const int stride = gridDim.x * 256;
    for (; i < n; i += stride) dst[i] = (_Float16)src[i];
}

// w_r2 [NR, PHID] f32 -> [16, PHID] f16, rows NR..15 zero (for WMMA phase-2)
__global__ __launch_bounds__(256)
void cvt_pad_w2_kernel(const float* __restrict__ src, _Float16* __restrict__ dst)
{
    const int i = blockIdx.x * 256 + threadIdx.x;
    if (i >= 16 * PHID_) return;
    const int n = i / PHID_;
    const int k = i % PHID_;
    dst[i] = (n < NR_) ? (_Float16)src[(size_t)n * PHID_ + k] : (_Float16)0.0f;
}

// =====================================================================
// Generic WMMA GEMM: C[M,N] = act(A[M,K] @ Wh[N,K]^T + bias[N])
// A: f32 (converted to f16 during LDS staging), Wh: pre-converted f16.
// Requires K % 32 == 0 (true for all call sites: 512/1024/2048).
// Tile BM=128, BN=128, BK=32. 256 thr = 8 waves; wave = 32x64 -> 8 accs.
// LDS rows padded to 48 halves (96B) so all segments are 16B aligned.
// =====================================================================
__global__ __launch_bounds__(256)
void gemm_bias_act(const float* __restrict__ A, const _Float16* __restrict__ Wh,
                   const float* __restrict__ bias, float* __restrict__ C,
                   int M, int N, int K, int act)
{
    const int PAD = 48;
    __shared__ _Float16 sA[128 * 48];
    __shared__ _Float16 sB[128 * 48];

    const int tid  = threadIdx.x;
    const int lane = tid & 31;
    const int wave = tid >> 5;
    const int wm   = wave >> 1;   // 0..3 : 32-row slab
    const int wn   = wave & 1;    // 0..1 : 64-col slab
    const int m0   = blockIdx.y * 128;
    const int n0   = blockIdx.x * 128;

    v8h hzero;
#pragma unroll
    for (int e = 0; e < 8; ++e) hzero[e] = (_Float16)0.0f;

    v8f acc[2][4];
#pragma unroll
    for (int mt = 0; mt < 2; ++mt)
#pragma unroll
        for (int nt = 0; nt < 4; ++nt)
#pragma unroll
            for (int r = 0; r < 8; ++r) acc[mt][nt][r] = 0.0f;

    const int lrow = tid >> 1;        // 0..127
    const int lk   = (tid & 1) * 16;  // 0 or 16
    const int hi   = (lane >> 4) & 1;
    const int l15  = lane & 15;

    for (int k0 = 0; k0 < K; k0 += 32) {
        // ---- stage A tile: 16 floats/thread as 4x float4, packed to 2x v8h
        {
            const int gm = m0 + lrow;
            _Float16* dst = sA + lrow * PAD + lk;
            if (gm < M) {
                const v4f* src = (const v4f*)(A + (size_t)gm * K + k0 + lk);
                v4f f0 = src[0], f1 = src[1], f2 = src[2], f3 = src[3];
                v8h h0, h1;
#pragma unroll
                for (int e = 0; e < 4; ++e) {
                    h0[e]     = (_Float16)f0[e];
                    h0[4 + e] = (_Float16)f1[e];
                    h1[e]     = (_Float16)f2[e];
                    h1[4 + e] = (_Float16)f3[e];
                }
                *(v8h*)(dst)     = h0;
                *(v8h*)(dst + 8) = h1;
            } else {
                *(v8h*)(dst)     = hzero;
                *(v8h*)(dst + 8) = hzero;
            }
        }
        // ---- stage B tile: pure 16B copies of pre-converted f16 weights
        for (int sgi = tid; sgi < 128 * 4; sgi += 256) {
            const int n  = sgi >> 2;
            const int sg = sgi & 3;
            const int gn = n0 + n;
            v8h val = hzero;
            if (gn < N) val = *(const v8h*)(Wh + (size_t)gn * K + k0 + sg * 8);
            *(v8h*)(sB + n * PAD + sg * 8) = val;
        }
        __syncthreads();

        // ---- fragments per ISA 7.12.2 layouts ----
        v16h afrag[2], bfrag[4];
#pragma unroll
        for (int mt = 0; mt < 2; ++mt) {
            const _Float16* pa = sA + (wm * 32 + mt * 16 + l15) * PAD;
            const int kb = hi * 8;
#pragma unroll
            for (int e = 0; e < 8; ++e) {
                afrag[mt][e]     = pa[kb + e];
                afrag[mt][8 + e] = pa[16 + kb + e];
            }
        }
#pragma unroll
        for (int nt = 0; nt < 4; ++nt) {
            const _Float16* pb = sB + (wn * 64 + nt * 16 + l15) * PAD + hi * 8;
#pragma unroll
            for (int e = 0; e < 8; ++e) { bfrag[nt][e] = pb[e]; bfrag[nt][8 + e] = pb[16 + e]; }
        }
#pragma unroll
        for (int mt = 0; mt < 2; ++mt)
#pragma unroll
            for (int nt = 0; nt < 4; ++nt)
                acc[mt][nt] = __builtin_amdgcn_wmma_f32_16x16x32_f16(
                    false, afrag[mt], false, bfrag[nt],
                    (short)0, acc[mt][nt], false, false);
        __syncthreads();
    }

    // ---- epilogue: bias + activation + store (ldc = N) ----
#pragma unroll
    for (int mt = 0; mt < 2; ++mt) {
#pragma unroll
        for (int nt = 0; nt < 4; ++nt) {
            const int gn = n0 + wn * 64 + nt * 16 + l15;
            if (gn >= N) continue;
            const float bv = bias ? bias[gn] : 0.0f;
#pragma unroll
            for (int r = 0; r < 8; ++r) {
                const int gm = m0 + wm * 32 + mt * 16 + hi * 8 + r;
                if (gm >= M) continue;
                float v = acc[mt][nt][r] + bv;
                if (act == 1) v = v > 0.0f ? v : 0.0f;
                C[(size_t)gm * N + gn] = v;
            }
        }
    }
}

// =====================================================================
// Attention helpers (~1% of total FLOPs -> plain VALU kernels)
// =====================================================================
__global__ __launch_bounds__(256)
void attn_scores_kernel(const float* __restrict__ qkv, float* __restrict__ attn)
{
    const int idx = blockIdx.x * 256 + threadIdx.x;
    if (idx >= B_ * H_ * S_ * S_) return;
    int r = idx;
    const int ki = r % S_; r /= S_;
    const int qi = r % S_; r /= S_;
    const int h  = r % H_;
    const int b  = r / H_;
    const float* q = qkv + (size_t)(b * S_ + qi) * (3 * D_) + h * DH_;
    const float* k = qkv + (size_t)(b * S_ + ki) * (3 * D_) + D_ + h * DH_;
    float s = 0.0f;
#pragma unroll 8
    for (int d = 0; d < DH_; ++d) s += q[d] * k[d];
    attn[idx] = s * 0.125f; // 1/sqrt(64)
}

__global__ __launch_bounds__(128)
void softmax_kernel(float* __restrict__ attn) // rows of length S_
{
    __shared__ float red[128];
    const int row = blockIdx.x;
    const int tid = threadIdx.x;
    float* a = attn + (size_t)row * S_;
    const int NPT = (S_ + 127) / 128; // 3
    float vals[NPT];
    float mx = -3.4e38f;
#pragma unroll
    for (int ii = 0; ii < NPT; ++ii) {
        const int c = tid + ii * 128;
        vals[ii] = (c < S_) ? a[c] : -3.4e38f;
        mx = fmaxf(mx, vals[ii]);
    }
    red[tid] = mx; __syncthreads();
    for (int off = 64; off > 0; off >>= 1) {
        if (tid < off) red[tid] = fmaxf(red[tid], red[tid + off]);
        __syncthreads();
    }
    mx = red[0]; __syncthreads();
    float s = 0.0f;
#pragma unroll
    for (int ii = 0; ii < NPT; ++ii) {
        const int c = tid + ii * 128;
        if (c < S_) { vals[ii] = __expf(vals[ii] - mx); s += vals[ii]; }
    }
    red[tid] = s; __syncthreads();
    for (int off = 64; off > 0; off >>= 1) {
        if (tid < off) red[tid] += red[tid + off];
        __syncthreads();
    }
    const float inv = 1.0f / red[0];
#pragma unroll
    for (int ii = 0; ii < NPT; ++ii) {
        const int c = tid + ii * 128;
        if (c < S_) a[c] = vals[ii] * inv;
    }
}

__global__ __launch_bounds__(256)
void attn_out_kernel(const float* __restrict__ attn, const float* __restrict__ qkv,
                     float* __restrict__ o)
{
    const int idx = blockIdx.x * 256 + threadIdx.x;
    if (idx >= B_ * S_ * D_) return;
    int r = idx;
    const int d = r % DH_; r /= DH_;
    const int h = r % H_;  r /= H_;
    const int s = r % S_;
    const int b = r / S_;
    const float* arow = attn + ((size_t)(b * H_ + h) * S_ + s) * S_;
    const float* v    = qkv + 2 * D_ + h * DH_ + d;
    float acc = 0.0f;
    for (int k = 0; k < S_; ++k)
        acc += arow[k] * v[(size_t)(b * S_ + k) * (3 * D_)];
    o[(size_t)(b * S_ + s) * D_ + h * DH_ + d] = acc;
}

// =====================================================================
// LayerNorm with optional residual add: Y = LN(X (+R)) * g + b
// =====================================================================
__global__ __launch_bounds__(256)
void ln_kernel(const float* __restrict__ X, const float* __restrict__ R,
               const float* __restrict__ g, const float* __restrict__ bta,
               float* __restrict__ Y, int cols)
{
    __shared__ float red[256];
    const int row = blockIdx.x;
    const int tid = threadIdx.x;
    const float* x = X + (size_t)row * cols;
    const float* rr = R ? R + (size_t)row * cols : nullptr;
    float vals[8];
    const int npt = (cols + 255) / 256; // <= 4 here
    float s = 0.0f;
    for (int ii = 0; ii < npt; ++ii) {
        const int c = tid + ii * 256;
        float v = 0.0f;
        if (c < cols) { v = x[c]; if (rr) v += rr[c]; }
        vals[ii] = v; s += v;
    }
    red[tid] = s; __syncthreads();
    for (int off = 128; off > 0; off >>= 1) {
        if (tid < off) red[tid] += red[tid + off];
        __syncthreads();
    }
    const float mean = red[0] / (float)cols;
    __syncthreads();
    float s2 = 0.0f;
    for (int ii = 0; ii < npt; ++ii) {
        const int c = tid + ii * 256;
        if (c < cols) { const float dd = vals[ii] - mean; s2 += dd * dd; }
    }
    red[tid] = s2; __syncthreads();
    for (int off = 128; off > 0; off >>= 1) {
        if (tid < off) red[tid] += red[tid + off];
        __syncthreads();
    }
    const float inv = rsqrtf(red[0] / (float)cols + 1e-5f);
    for (int ii = 0; ii < npt; ++ii) {
        const int c = tid + ii * 256;
        if (c < cols)
            Y[(size_t)row * cols + c] = (vals[ii] - mean) * inv * g[c] + bta[c];
    }
}

// =====================================================================
// Fused pair relation classifier (dominant: ~106 of ~120 GFLOP).
//   feat[p] = [ch[b,i], ch[b,j], de_h[bucket]]  (f16 gather, 16B segments)
//   h = leaky_relu(feat @ w_r1^T + b_r1)        WMMA, K=576, N=512
//   rel[p] = h @ w_r2^T + b_r2                  WMMA (N padded 10->16),
//                                               8-way cross-wave K reduce
// PM=32 pair rows / block; 8 waves each own 64 hidden cols in phase 1,
// and a 64-wide K-slice in phase 2.
// B tile staged with global_load_async_to_lds_b128 (ASYNCcnt path).
// =====================================================================
__global__ __launch_bounds__(256)
void pair_rel_kernel(const _Float16* __restrict__ ch,      // [B,S,DS] f16
                     const int*      __restrict__ pos,     // [B,S]
                     const _Float16* __restrict__ wr1h,    // [PHID, F] f16
                     const float*    __restrict__ b_r1,    // [PHID]
                     const _Float16* __restrict__ wr2h,    // [16, PHID] f16 padded
                     const float*    __restrict__ b_r2,    // [NR]
                     const _Float16* __restrict__ de_h,    // [6, DE] f16
                     float*          __restrict__ out)     // [P, NR]
{
    const int PM = 32, PKP = 48, PNP = 520;
    __shared__ union {
        struct { _Float16 A[32 * 48]; _Float16 Bm[512 * 48]; } g;    // 52.2 KB
        struct { _Float16 Hm[32 * 520]; float R2[8 * 32 * 16]; } h2; // 49.7 KB
    } sm;
    __shared__ int rowOffA[PM], rowOffB[PM], rowOffD[PM]; // per-row gather bases

    const int tid   = threadIdx.x;
    const int lane  = tid & 31;
    const int wave  = tid >> 5;          // phase1: 64-col slab; phase2: K slice
    const int hi    = (lane >> 4) & 1;
    const int l15   = lane & 15;
    const int pbase = blockIdx.x * PM;

    // ---- decode pair rows once per block ----
    if (tid < PM) {
        const int p = pbase + tid;
        int oa = -1, ob = 0, od = 0;
        if (p < P_) {
            const unsigned q = (unsigned)p;
            const unsigned b = q / (unsigned)PP_;
            const unsigned w = q % (unsigned)PP_;
            const unsigned i = w / (unsigned)(S_ - 1);
            const unsigned t = w % (unsigned)(S_ - 1);
            const unsigned j = t + (t >= i ? 1u : 0u);
            oa = (int)((b * S_ + i) * DS_);
            ob = (int)((b * S_ + j) * DS_);
            int d = pos[b * S_ + i] - pos[b * S_ + j];
            d = d < 0 ? -d : d;
            const int bk = (d > 5) + (d > 10) + (d > 20) + (d > 50) + (d > 100);
            od = bk * DE_;
        }
        rowOffA[tid] = oa; rowOffB[tid] = ob; rowOffD[tid] = od;
    }

    v8h hzero;
#pragma unroll
    for (int e = 0; e < 8; ++e) hzero[e] = (_Float16)0.0f;

    v8f acc[2][4];
#pragma unroll
    for (int mt = 0; mt < 2; ++mt)
#pragma unroll
        for (int nt = 0; nt < 4; ++nt)
#pragma unroll
            for (int r = 0; r < 8; ++r) acc[mt][nt][r] = 0.0f;

    __syncthreads(); // row descriptors visible

    // ================= phase 1: h = feat @ w_r1^T =================
    for (int k0 = 0; k0 < F_; k0 += 32) {
        // stage B tile via async global->LDS 16B copies (8 per thread)
        for (int sgi = tid; sgi < PHID_ * 4; sgi += 256) {
            const int n  = sgi >> 2;
            const int sg = sgi & 3;
            async_copy_b128(wr1h + (size_t)n * F_ + k0 + sg * 8,
                            sm.g.Bm + n * PKP + sg * 8);
        }
        // stage A tile: gathered 16B segments (DS_, 2*DS_, F_ are multiples
        // of 8 so a segment never straddles source tensors)
        for (int sgi = tid; sgi < PM * 4; sgi += 256) {
            const int r  = sgi >> 2;
            const int sg = sgi & 3;
            const int f  = k0 + sg * 8;
            const int oa = rowOffA[r];
            v8h val = hzero;
            if (oa >= 0) {
                const _Float16* srcp =
                    (f < DS_)     ? ch + oa + f :
                    (f < 2 * DS_) ? ch + rowOffB[r] + (f - DS_) :
                                    de_h + rowOffD[r] + (f - 2 * DS_);
                val = *(const v8h*)srcp;
            }
            *(v8h*)(sm.g.A + r * PKP + sg * 8) = val;
        }
        wait_async0();          // B-tile async copies complete before barrier
        __syncthreads();

        v16h afrag[2], bfrag[4];
#pragma unroll
        for (int mt = 0; mt < 2; ++mt) {
            const _Float16* pa = sm.g.A + (mt * 16 + l15) * PKP;
            const int kb = hi * 8;
#pragma unroll
            for (int e = 0; e < 8; ++e) {
                afrag[mt][e]     = pa[kb + e];
                afrag[mt][8 + e] = pa[16 + kb + e];
            }
        }
#pragma unroll
        for (int nt = 0; nt < 4; ++nt) {
            const _Float16* pb = sm.g.Bm + (wave * 64 + nt * 16 + l15) * PKP + hi * 8;
#pragma unroll
            for (int e = 0; e < 8; ++e) { bfrag[nt][e] = pb[e]; bfrag[nt][8 + e] = pb[16 + e]; }
        }
#pragma unroll
        for (int mt = 0; mt < 2; ++mt)
#pragma unroll
            for (int nt = 0; nt < 4; ++nt)
                acc[mt][nt] = __builtin_amdgcn_wmma_f32_16x16x32_f16(
                    false, afrag[mt], false, bfrag[nt],
                    (short)0, acc[mt][nt], false, false);
        __syncthreads();
    }

    // ---- bias + leaky_relu -> H tile (f16) in the reused LDS slab ----
#pragma unroll
    for (int mt = 0; mt < 2; ++mt) {
#pragma unroll
        for (int nt = 0; nt < 4; ++nt) {
            const int col = wave * 64 + nt * 16 + l15;
            const float bv = b_r1[col];
#pragma unroll
            for (int r = 0; r < 8; ++r) {
                const int row = mt * 16 + hi * 8 + r;
                float v = acc[mt][nt][r] + bv;
                v = v > 0.0f ? v : 0.01f * v;
                sm.h2.Hm[row * PNP + col] = (_Float16)v;
            }
        }
    }
    __syncthreads();

    // ========== phase 2: rel = H @ w_r2^T on the matrix pipe ==========
    // wave handles K in [wave*64, wave*64+64); partials reduced via LDS.
    v8f acc2[2];
#pragma unroll
    for (int mt = 0; mt < 2; ++mt)
#pragma unroll
        for (int r = 0; r < 8; ++r) acc2[mt][r] = 0.0f;

#pragma unroll
    for (int kc = 0; kc < 2; ++kc) {
        const int k0 = wave * 64 + kc * 32;
        v16h af[2], bf;
#pragma unroll
        for (int mt = 0; mt < 2; ++mt) {
            const _Float16* pa = sm.h2.Hm + (mt * 16 + l15) * PNP + k0;
            const int kb = hi * 8;
#pragma unroll
            for (int e = 0; e < 8; ++e) {
                af[mt][e]     = pa[kb + e];
                af[mt][8 + e] = pa[16 + kb + e];
            }
        }
        // B fragment straight from 16KB L2-resident padded w_r2 (lane = out col)
        {
            const _Float16* pb = wr2h + (size_t)l15 * PHID_ + k0 + hi * 8;
            const v8h b0 = *(const v8h*)(pb);
            const v8h b1 = *(const v8h*)(pb + 16);
#pragma unroll
            for (int e = 0; e < 8; ++e) { bf[e] = b0[e]; bf[8 + e] = b1[e]; }
        }
#pragma unroll
        for (int mt = 0; mt < 2; ++mt)
            acc2[mt] = __builtin_amdgcn_wmma_f32_16x16x32_f16(
                false, af[mt], false, bf, (short)0, acc2[mt], false, false);
    }
    // write this wave's 32x16 partial
#pragma unroll
    for (int mt = 0; mt < 2; ++mt)
#pragma unroll
        for (int r = 0; r < 8; ++r) {
            const int row = mt * 16 + hi * 8 + r;
            sm.h2.R2[(wave * 32 + row) * 16 + l15] = acc2[mt][r];
        }
    __syncthreads();

    // ---- 8-way cross-wave reduction + bias + store ----
    for (int idx = tid; idx < PM * 16; idx += 256) {
        const int row = idx >> 4;
        const int col = idx & 15;
        if (col >= NR_) continue;
        const int p = pbase + row;
        if (p >= P_) continue;
        float s = 0.0f;
#pragma unroll
        for (int w = 0; w < 8; ++w) s += sm.h2.R2[(w * 32 + row) * 16 + col];
        out[(size_t)p * NR_ + col] = s + b_r2[col];
    }
}

// =====================================================================
// Host-side orchestration
// =====================================================================
static inline void launch_gemm(const float* A, const _Float16* Wh, const float* bias,
                               float* C, int M, int N, int K, int act,
                               hipStream_t stream)
{
    dim3 grid((N + 127) / 128, (M + 127) / 128);
    gemm_bias_act<<<grid, dim3(256), 0, stream>>>(A, Wh, bias, C, M, N, K, act);
}

static inline void launch_cvt(const float* s, _Float16* d, int n, hipStream_t stream)
{
    int blocks = (n + 1023) / 1024;
    if (blocks < 1) blocks = 1;
    if (blocks > 2048) blocks = 2048;
    cvt_f32_f16_kernel<<<blocks, 256, 0, stream>>>(s, d, n);
}

extern "C" void kernel_launch(void* const* d_in, const int* in_sizes, int n_in,
                              void* d_out, int out_size, void* d_ws, size_t ws_size,
                              hipStream_t stream)
{
    (void)in_sizes; (void)n_in; (void)out_size; (void)ws_size;

    // ---- inputs (setup_inputs dict order, params flattened in insertion order)
    const float* span_repr = (const float*)d_in[0];
    const int*   span_pos  = (const int*)  d_in[1];
    const float* w_qkv  = (const float*)d_in[2];   const float* b_qkv  = (const float*)d_in[3];
    const float* w_o    = (const float*)d_in[4];   const float* b_o    = (const float*)d_in[5];
    const float* ln1_g  = (const float*)d_in[6];   const float* ln1_b  = (const float*)d_in[7];
    const float* w_ff1  = (const float*)d_in[8];   const float* b_ff1  = (const float*)d_in[9];
    const float* w_ff2  = (const float*)d_in[10];  const float* b_ff2  = (const float*)d_in[11];
    const float* ln2_g  = (const float*)d_in[12];  const float* ln2_b  = (const float*)d_in[13];
    const float* reg_g  = (const float*)d_in[14];  const float* reg_b  = (const float*)d_in[15];
    const float* w_et   = (const float*)d_in[16];  const float* b_et   = (const float*)d_in[17];
    const float* w_tt   = (const float*)d_in[18];  const float* b_tt   = (const float*)d_in[19];
    const float* w_c1   = (const float*)d_in[20];  const float* b_c1   = (const float*)d_in[21];
    const float* c_ln1g = (const float*)d_in[22];  const float* c_ln1b = (const float*)d_in[23];
    const float* w_c2   = (const float*)d_in[24];  const float* b_c2   = (const float*)d_in[25];
    const float* c_ln2g = (const float*)d_in[26];  const float* c_ln2b = (const float*)d_in[27];
    const float* w_r1   = (const float*)d_in[28];  const float* b_r1   = (const float*)d_in[29];
    const float* w_r2   = (const float*)d_in[30];  const float* b_r2   = (const float*)d_in[31];
    const float* dist_e = (const float*)d_in[32];

    // ---- workspace: f32 activations then f16 mirrors (all 16B aligned)
    float* ws   = (float*)d_ws;
    size_t o = 0;
    float* x    = ws + o; o += (size_t)MR_ * D_;          // [600,512]
    float* big  = ws + o; o += (size_t)MR_ * FF_;         // [600,3D]/[600,FF]
    float* attn = ws + o; o += (size_t)B_ * H_ * S_ * S_; // [B,H,S,S]
    float* obuf = ws + o; o += (size_t)MR_ * D_;
    float* tbuf = ws + o; o += (size_t)MR_ * D_;
    float* c1   = ws + o; o += (size_t)MR_ * 2 * D_;
    float* cbuf = ws + o; o += (size_t)MR_ * DS_;

    _Float16* hp = (_Float16*)(ws + o);
    _Float16* wqkv_h = hp; hp += (size_t)L_ * 3 * D_ * D_;
    _Float16* wo_h   = hp; hp += (size_t)L_ * D_ * D_;
    _Float16* wf1_h  = hp; hp += (size_t)L_ * FF_ * D_;
    _Float16* wf2_h  = hp; hp += (size_t)L_ * D_ * FF_;
    _Float16* wc1_h  = hp; hp += (size_t)2 * D_ * D_;
    _Float16* wc2_h  = hp; hp += (size_t)DS_ * 2 * D_;
    _Float16* wet_h  = hp; hp += (size_t)NE_ * D_;
    _Float16* wtt_h  = hp; hp += (size_t)NT_ * D_;
    _Float16* wr1_h  = hp; hp += (size_t)PHID_ * F_;
    _Float16* wr2_h  = hp; hp += (size_t)16 * PHID_;      // padded 10->16 rows
    _Float16* ch     = hp; hp += (size_t)MR_ * DS_;
    _Float16* de_h   = hp; hp += (size_t)6 * DE_;

    float* out_ent = (float*)d_out;                 // [600, 20]
    float* out_trg = out_ent + (size_t)MR_ * NE_;   // [600, 34]
    float* out_rel = out_trg + (size_t)MR_ * NT_;   // [P, 10]

    // seed activation x = span_repr (D2D async copy is graph-capture safe)
    hipMemcpyAsync(x, span_repr, (size_t)MR_ * D_ * sizeof(float),
                   hipMemcpyDeviceToDevice, stream);

    // ---- one-shot weight conversions to f16 ----
    launch_cvt(w_qkv, wqkv_h, L_ * 3 * D_ * D_, stream);
    launch_cvt(w_o,   wo_h,   L_ * D_ * D_,     stream);
    launch_cvt(w_ff1, wf1_h,  L_ * FF_ * D_,    stream);
    launch_cvt(w_ff2, wf2_h,  L_ * D_ * FF_,    stream);
    launch_cvt(w_c1,  wc1_h,  2 * D_ * D_,      stream);
    launch_cvt(w_c2,  wc2_h,  DS_ * 2 * D_,     stream);
    launch_cvt(w_et,  wet_h,  NE_ * D_,         stream);
    launch_cvt(w_tt,  wtt_h,  NT_ * D_,         stream);
    launch_cvt(w_r1,  wr1_h,  PHID_ * F_,       stream);
    launch_cvt(dist_e, de_h,  6 * DE_,          stream);
    cvt_pad_w2_kernel<<<(16 * PHID_ + 255) / 256, 256, 0, stream>>>(w_r2, wr2_h);

    // ---- 3 transformer encoder layers ----
    for (int i = 0; i < L_; ++i) {
        launch_gemm(x, wqkv_h + (size_t)i * 3 * D_ * D_, b_qkv + (size_t)i * 3 * D_,
                    big, MR_, 3 * D_, D_, 0, stream);
        attn_scores_kernel<<<(B_ * H_ * S_ * S_ + 255) / 256, 256, 0, stream>>>(big, attn);
        softmax_kernel<<<B_ * H_ * S_, 128, 0, stream>>>(attn);
        attn_out_kernel<<<(MR_ * D_ + 255) / 256, 256, 0, stream>>>(attn, big, obuf);
        launch_gemm(obuf, wo_h + (size_t)i * D_ * D_, b_o + (size_t)i * D_,
                    tbuf, MR_, D_, D_, 0, stream);
        ln_kernel<<<MR_, 256, 0, stream>>>(x, tbuf, ln1_g + (size_t)i * D_,
                                           ln1_b + (size_t)i * D_, x, D_);
        launch_gemm(x, wf1_h + (size_t)i * FF_ * D_, b_ff1 + (size_t)i * FF_,
                    big, MR_, FF_, D_, 1, stream);  // relu
        launch_gemm(big, wf2_h + (size_t)i * D_ * FF_, b_ff2 + (size_t)i * D_,
                    tbuf, MR_, D_, FF_, 0, stream);
        ln_kernel<<<MR_, 256, 0, stream>>>(x, tbuf, ln2_g + (size_t)i * D_,
                                           ln2_b + (size_t)i * D_, x, D_);
        ln_kernel<<<MR_, 256, 0, stream>>>(x, nullptr, reg_g + (size_t)i * D_,
                                           reg_b + (size_t)i * D_, x, D_);
    }

    // ---- entity / trigger type classifiers ----
    launch_gemm(x, wet_h, b_et, out_ent, MR_, NE_, D_, 0, stream);
    launch_gemm(x, wtt_h, b_tt, out_trg, MR_, NT_, D_, 0, stream);

    // ---- before_rel_compress: Linear -> LN -> Linear -> LN ----
    launch_gemm(x, wc1_h, b_c1, c1, MR_, 2 * D_, D_, 0, stream);
    ln_kernel<<<MR_, 256, 0, stream>>>(c1, nullptr, c_ln1g, c_ln1b, c1, 2 * D_);
    launch_gemm(c1, wc2_h, b_c2, cbuf, MR_, DS_, 2 * D_, 0, stream);
    ln_kernel<<<MR_, 256, 0, stream>>>(cbuf, nullptr, c_ln2g, c_ln2b, cbuf, DS_);

    // compressed reps + dist embeddings to f16 for the pair gather
    launch_cvt(cbuf, ch, MR_ * DS_, stream);

    // ---- fused pair relation classifier (dominant GEMM, WMMA + async LDS) ----
    pair_rel_kernel<<<(P_ + 31) / 32, 256, 0, stream>>>(
        ch, span_pos, wr1_h, b_r1, wr2_h, b_r2, de_h, out_rel);
}
